// MoE_82617990905868
// MI455X (gfx1250) — compile-verified, matrix operand order checked by reading, not run on hardware
//
#include <hip/hip_runtime.h>
#include <math.h>

// ---- problem constants (B=4,S=2048,D=1024,E=8,H=2048,top_k=2) ----
#define T_TOK 8192
#define D_DIM 1024
#define E_EXP 8
#define H_DIM 2048
#define TILES 512          // T_TOK / 16 token-tiles per expert (worst case)

typedef __bf16 bf16;
typedef __attribute__((ext_vector_type(16))) __bf16 v16bf;
typedef __attribute__((ext_vector_type(8)))  float  v8f;

union Frag { uint4 q[2]; v16bf v; };   // 32 bytes = one 16-bit WMMA operand

// padded LDS strides (elements) to spread rows across banks
#define XG_STR 1032        // 16 x 1024 bf16 activation tile
#define HT_STR 136         // 16 x 128  bf16 h tile

// generic pointer -> 32-bit LDS offset (ISA 10.2: LDS aperture addr[31:0])
__device__ __forceinline__ unsigned lds_off(const void* p) {
    return (unsigned)(unsigned long long)p;
}

// ------------------------------------------------------------------
// 0) zero output + expert counters
__global__ void moe_zero_kernel(float* __restrict__ out, int n, int* __restrict__ counts) {
    int i = blockIdx.x * blockDim.x + threadIdx.x;
    if (i < n) out[i] = 0.f;
    if (i < E_EXP) counts[i] = 0;
}

// 1) x (f32) -> xb (bf16), pairwise to encourage v_cvt_pk_bf16_f32
__global__ void moe_convx_kernel(const float* __restrict__ x, bf16* __restrict__ xb, int n2) {
    int i = blockIdx.x * blockDim.x + threadIdx.x;
    if (i >= n2) return;
    float2 v = ((const float2*)x)[i];
    xb[2*i + 0] = (bf16)v.x;
    xb[2*i + 1] = (bf16)v.y;
}

// 2) W1 [E,D,H] f32 -> W1bt [E,H,D] bf16 (transposed: contiguous in K=D)
__global__ void moe_convw1_kernel(const float* __restrict__ W1, bf16* __restrict__ W1bt) {
    long i = (long)blockIdx.x * blockDim.x + threadIdx.x;
    if (i >= (long)E_EXP * H_DIM * D_DIM) return;
    int  d = (int)(i % D_DIM);
    long r = i / D_DIM;
    int  h = (int)(r % H_DIM);
    int  e = (int)(r / H_DIM);
    W1bt[i] = (bf16)W1[((long)e * D_DIM + d) * H_DIM + h];
}

// 3) W2 [E,H,D] f32 -> W2bt [E,D,H] bf16 (transposed: contiguous in K=H)
__global__ void moe_convw2_kernel(const float* __restrict__ W2, bf16* __restrict__ W2bt) {
    long i = (long)blockIdx.x * blockDim.x + threadIdx.x;
    if (i >= (long)E_EXP * D_DIM * H_DIM) return;
    int  h = (int)(i % H_DIM);
    long r = i / H_DIM;
    int  d = (int)(r % D_DIM);
    int  e = (int)(r / D_DIM);
    W2bt[i] = (bf16)W2[((long)e * H_DIM + h) * D_DIM + d];
}

// 4) gating: one wave per token -> combine weights w[t][E] (0 except top-2, renormalized)
__global__ void moe_gate_kernel(const float* __restrict__ x,
                                const float* __restrict__ Wg,
                                const float* __restrict__ bg,
                                float* __restrict__ wout) {
    const int wave = threadIdx.x >> 5;
    const int lane = threadIdx.x & 31;
    const int t = blockIdx.x * (blockDim.x >> 5) + wave;
    if (t >= T_TOK) return;

    float acc[E_EXP] = {0.f,0.f,0.f,0.f,0.f,0.f,0.f,0.f};
    const float* xr = x + (long)t * D_DIM;
    for (int d = lane; d < D_DIM; d += 32) {
        float xv = xr[d];
        #pragma unroll
        for (int e = 0; e < E_EXP; ++e) acc[e] += xv * Wg[d * E_EXP + e];
    }
    #pragma unroll
    for (int e = 0; e < E_EXP; ++e) {
        #pragma unroll
        for (int m = 16; m > 0; m >>= 1) acc[e] += __shfl_xor(acc[e], m, 32);
    }
    if (lane == 0) {
        float lg[E_EXP], mx = -1e30f;
        #pragma unroll
        for (int e = 0; e < E_EXP; ++e) { lg[e] = acc[e] + bg[e]; mx = fmaxf(mx, lg[e]); }
        float ex[E_EXP];
        #pragma unroll
        for (int e = 0; e < E_EXP; ++e) ex[e] = __expf(lg[e] - mx);
        // top-2 on exp-logits (softmax denom cancels in renormalization)
        int i1 = 0;
        #pragma unroll
        for (int e = 1; e < E_EXP; ++e) if (ex[e] > ex[i1]) i1 = e;
        int i2 = (i1 == 0) ? 1 : 0;
        #pragma unroll
        for (int e = 0; e < E_EXP; ++e) if (e != i1 && ex[e] > ex[i2]) i2 = e;
        float s = ex[i1] + ex[i2];
        #pragma unroll
        for (int e = 0; e < E_EXP; ++e)
            wout[(long)t * E_EXP + e] = (e == i1) ? ex[i1] / s : ((e == i2) ? ex[i2] / s : 0.f);
    }
}

// 5) routing: append each token to its 2 experts' lists
__global__ void moe_route_kernel(const float* __restrict__ wcomb,
                                 int* __restrict__ counts,
                                 int* __restrict__ toklist,
                                 float* __restrict__ wtlist) {
    int t = blockIdx.x * blockDim.x + threadIdx.x;
    if (t >= T_TOK) return;
    #pragma unroll
    for (int e = 0; e < E_EXP; ++e) {
        float w = wcomb[(long)t * E_EXP + e];
        if (w > 0.f) {
            int pos = atomicAdd(&counts[e], 1);
            toklist[e * T_TOK + pos] = t;
            wtlist[e * T_TOK + pos] = w;
        }
    }
}

// 6) grouped FFN: one workgroup = (expert, 16-token tile), 8 waves, bf16 WMMA
__launch_bounds__(256)
__global__ void moe_ffn_kernel(const bf16* __restrict__ xb,
                               const bf16* __restrict__ W1bt,
                               const bf16* __restrict__ W2bt,
                               const float* __restrict__ b1,
                               const float* __restrict__ b2,
                               const int*  __restrict__ counts,
                               const int*  __restrict__ toklist,
                               const float* __restrict__ wtlist,
                               float* __restrict__ out) {
    const int e    = blockIdx.x / TILES;
    const int tile = blockIdx.x % TILES;
    const int cnt  = counts[e];
    const int start = tile * 16;
    if (start >= cnt) return;           // uniform early-exit

    __shared__ __align__(16) bf16 xg[16 * XG_STR];
    __shared__ __align__(16) bf16 ht[16 * HT_STR];
    __shared__ int   toks[16];
    __shared__ float wts[16];

    const int tid  = threadIdx.x;
    const int wave = tid >> 5;
    const int lane = tid & 31;
    const int half = lane >> 4;         // 0: lanes 0-15, 1: lanes 16-31
    const int l15  = lane & 15;

    if (tid < 16) {
        int idx = start + tid;
        if (idx < cnt) { toks[tid] = toklist[e * T_TOK + idx]; wts[tid] = wtlist[e * T_TOK + idx]; }
        else           { toks[tid] = 0;                        wts[tid] = 0.f; }   // padded rows weigh 0
    }
    __syncthreads();

    // ---- stage 16 gathered token rows into LDS via async memory->LDS DMA ----
    // global_load_async_to_lds_b128: per-lane global addr (VGPR pair) -> per-lane
    // LDS addr (VGPR), tracked by ASYNCcnt (ISA 10.x / CDNA5 ISA).
    {
        const unsigned xg_base = lds_off(&xg[0]);
        for (int i = tid; i < 16 * 128; i += 256) {          // 128 x 16B chunks per row
            int row = i >> 7, off8 = (i & 127) * 8;
            unsigned laddr = xg_base + (unsigned)(row * XG_STR + off8) * 2u;
            unsigned long long gaddr =
                (unsigned long long)&xb[(size_t)toks[row] * D_DIM + off8];
            asm volatile("global_load_async_to_lds_b128 %0, %1, off"
                         :: "v"(laddr), "v"(gaddr) : "memory");
        }
        asm volatile("s_wait_asynccnt 0x0" ::: "memory");
    }
    __syncthreads();

    const v8f vzero = {};
    v8f acc[8];                         // this wave's 16 x 128 slice of y (f32)
    #pragma unroll
    for (int t = 0; t < 8; ++t) acc[t] = vzero;

    for (int hc = 0; hc < H_DIM; hc += 128) {
        const int ncol = hc + wave * 16 + l15;                    // this lane's h column
        const bf16* w1row = W1bt + ((size_t)e * H_DIM + ncol) * D_DIM;

        // warm L2/L0 for this chunk's W2 rows (consumed in GEMM2 below)
        #pragma unroll
        for (int t = 0; t < 8; ++t) {
            const int dcol = wave * 128 + t * 16 + l15;
            __builtin_prefetch(W2bt + ((size_t)e * D_DIM + dcol) * H_DIM + hc, 0, 0);
        }
        // warm next chunk's W1 row
        if (hc + 128 < H_DIM)
            __builtin_prefetch(W1bt + ((size_t)e * H_DIM + (ncol + 128)) * D_DIM, 0, 0);

        // ---- GEMM1: h[:, hc + wave*16 .. +16) = x(16xD) * W1, K = D = 1024 ----
        v8f hacc = vzero;
        for (int k0 = 0; k0 < D_DIM; k0 += 32) {
            Frag a, b;
            // A fragment: ISA 16-bit A layout -> two b128 LDS loads per lane
            a.q[0] = *(const uint4*)&xg[l15 * XG_STR + k0 + half * 8];
            a.q[1] = *(const uint4*)&xg[l15 * XG_STR + k0 + 16 + half * 8];
            // B fragment: transposed weights -> contiguous K, two b128 global loads
            b.q[0] = *(const uint4*)&w1row[k0 + half * 8];
            b.q[1] = *(const uint4*)&w1row[k0 + 16 + half * 8];
            hacc = __builtin_amdgcn_wmma_f32_16x16x32_bf16(
                       false, a.v, false, b.v, (short)0, hacc, false, false);
        }
        // epilogue: +b1, tanh-gelu, store bf16 h tile to LDS (row-major, padded)
        const float b1v = b1[(size_t)e * H_DIM + ncol];
        #pragma unroll
        for (int r = 0; r < 8; ++r) {
            float h = hacc[r] + b1v;                              // row M = r + half*8
            float g = 0.5f * h * (1.f + tanhf(0.7978845608f * (h + 0.044715f * h * h * h)));
            ht[(r + half * 8) * HT_STR + wave * 16 + l15] = (bf16)g;
        }
        __syncthreads();

        // ---- GEMM2: y(16x1024) += h_tile(16x128) * W2, this wave owns 128 cols ----
        #pragma unroll
        for (int kk = 0; kk < 128; kk += 32) {
            Frag a;
            a.q[0] = *(const uint4*)&ht[l15 * HT_STR + kk + half * 8];
            a.q[1] = *(const uint4*)&ht[l15 * HT_STR + kk + 16 + half * 8];
            #pragma unroll
            for (int t = 0; t < 8; ++t) {
                const int dcol = wave * 128 + t * 16 + l15;
                const bf16* w2row = W2bt + ((size_t)e * D_DIM + dcol) * H_DIM + hc;
                Frag b;
                b.q[0] = *(const uint4*)&w2row[kk + half * 8];
                b.q[1] = *(const uint4*)&w2row[kk + 16 + half * 8];
                acc[t] = __builtin_amdgcn_wmma_f32_16x16x32_bf16(
                             false, a.v, false, b.v, (short)0, acc[t], false, false);
            }
        }
        __syncthreads();
    }

    // ---- epilogue: out[tok] += w * (y + b2), scatter via f32 atomics ----
    #pragma unroll
    for (int t = 0; t < 8; ++t) {
        const int dcol = wave * 128 + t * 16 + l15;
        const float b2v = b2[(size_t)e * D_DIM + dcol];
        #pragma unroll
        for (int r = 0; r < 8; ++r) {
            const int m = r + half * 8;                            // C-layout row
            float val = (acc[t][r] + b2v) * wts[m];
            atomicAdd(&out[(size_t)toks[m] * D_DIM + dcol], val);
        }
    }
}

// ------------------------------------------------------------------
extern "C" void kernel_launch(void* const* d_in, const int* in_sizes, int n_in,
                              void* d_out, int out_size, void* d_ws, size_t ws_size,
                              hipStream_t stream) {
    const float* x  = (const float*)d_in[0];
    // d_in[1] = top_k scalar (fixed 2)
    const float* Wg = (const float*)d_in[2];
    const float* bg = (const float*)d_in[3];
    const float* W1 = (const float*)d_in[4];
    const float* b1 = (const float*)d_in[5];
    const float* W2 = (const float*)d_in[6];
    const float* b2 = (const float*)d_in[7];
    float* out = (float*)d_out;

    // workspace layout
    char* ws = (char*)d_ws;
    float* wcomb   = (float*)(ws);                       // T*E f32      (256 KB)
    int*   counts  = (int*)  (ws + (256u << 10));        // E ints (padded to 1 KB)
    int*   toklist = (int*)  (ws + (256u << 10) + 1024); // E*T ints     (256 KB)
    float* wtlist  = (float*)(ws + (512u << 10) + 1024); // E*T f32      (256 KB)
    size_t off = ((768u << 10) + 1024 + 4095) & ~(size_t)4095;
    bf16* xb   = (bf16*)(ws + off);                      // T*D bf16     (16 MB)
    off += (size_t)T_TOK * D_DIM * 2;
    bf16* W1bt = (bf16*)(ws + off);                      // E*H*D bf16   (32 MB)
    off += (size_t)E_EXP * H_DIM * D_DIM * 2;
    bf16* W2bt = (bf16*)(ws + off);                      // E*D*H bf16   (32 MB)

    const int outN = T_TOK * D_DIM;                      // == out_size
    moe_zero_kernel<<<(outN + 255) / 256, 256, 0, stream>>>(out, outN, counts);

    const int n2 = (T_TOK * D_DIM) / 2;
    moe_convx_kernel<<<(n2 + 255) / 256, 256, 0, stream>>>(x, xb, n2);

    const long nw = (long)E_EXP * H_DIM * D_DIM;         // 16.8M
    moe_convw1_kernel<<<(unsigned)((nw + 255) / 256), 256, 0, stream>>>(W1, W1bt);
    moe_convw2_kernel<<<(unsigned)((nw + 255) / 256), 256, 0, stream>>>(W2, W2bt);

    moe_gate_kernel<<<T_TOK / 8, 256, 0, stream>>>(x, Wg, bg, wcomb);
    moe_route_kernel<<<T_TOK / 256, 256, 0, stream>>>(wcomb, counts, toklist, wtlist);

    moe_ffn_kernel<<<E_EXP * TILES, 256, 0, stream>>>(xb, W1bt, W2bt, b1, b2,
                                                      counts, toklist, wtlist, out);
}
